// RPN_80384607912361
// MI455X (gfx1250) — matrix operand order repaired and checked by reading, not compiled, not run on hardware
//
#include <hip/hip_runtime.h>
#include <hip/hip_bf16.h>

typedef __attribute__((ext_vector_type(16))) _Float16 v16h;
typedef __attribute__((ext_vector_type(8)))  float    v8f;

#define Bn   8
#define Cin  512
#define Hh   64
#define Ww   64
#define OC   512
#define K9   4608          // 512*9
#define NPROP 36864        // 64*64*9
#define PRE  6000
#define POST 300

// base anchor widths/heights (ratio-major, scale-minor) from _generate_base_anchors
__device__ __constant__ float c_AW[9] = {184.f,368.f,736.f,128.f,256.f,512.f, 88.f,176.f,352.f};
__device__ __constant__ float c_AH[9] = { 96.f,192.f,384.f,128.f,256.f,512.f,176.f,352.f,704.f};

// ---------------- CDNA5 async global->LDS helpers ----------------
// global_load_async_to_lds_b128: per-lane 16B DMA into LDS, tracked by ASYNCcnt.
// LDS operand is the low 32 bits of the generic pointer (flat LDS aperture maps
// addr[31:0] -> LDS offset).
__device__ __forceinline__ void async_copy_b128(void* lds_dst, const void* gsrc) {
    unsigned loff = (unsigned)(size_t)lds_dst;
    asm volatile("global_load_async_to_lds_b128 %0, %1, off"
                 :: "v"(loff), "v"(gsrc) : "memory");
}
__device__ __forceinline__ void async_wait0() {
    asm volatile("s_wait_asynccnt 0" ::: "memory");
}

// ---------------- staging kernels ----------------
__global__ void rpn_cvt_input(const float* __restrict__ x, _Float16* __restrict__ y, int n) {
    int i = blockIdx.x * blockDim.x + threadIdx.x;
    if (i < n) y[i] = (_Float16)x[i];
}

// w_conv [oc][ic][3][3] -> W16 [oc][pos][ic]  (pos = ky*3+kx)
__global__ void rpn_cvt_wconv(const float* __restrict__ w, _Float16* __restrict__ o) {
    int gid = blockIdx.x * blockDim.x + threadIdx.x;
    if (gid >= OC * K9) return;
    int oc = gid / K9;
    int r  = gid % K9;
    int pos = r / Cin, ic = r % Cin;
    int ky = pos / 3, kx = pos % 3;
    o[gid] = (_Float16)w[((oc * Cin + ic) * 3 + ky) * 3 + kx];
}

// pack cls(18) + bbox(36) 1x1 weights into [64][512] (rows 54..63 zero)
__global__ void rpn_cvt_w2(const float* __restrict__ wc, const float* __restrict__ wb,
                           _Float16* __restrict__ o) {
    int gid = blockIdx.x * blockDim.x + threadIdx.x;
    if (gid >= 64 * Cin) return;
    int r = gid / Cin, ic = gid % Cin;
    float v = 0.f;
    if (r < 18)      v = wc[r * Cin + ic];
    else if (r < 54) v = wb[(r - 18) * Cin + ic];
    o[gid] = (_Float16)v;
}

// ---------------- WMMA fragment layout ----------------
// A (16x32 f16): lane l -> M=l&15, hsel=l>>4; halves 0..7 = K hsel*8+0..7, halves 8..15 = K 16+hsel*8+0..7
// B (32x16 f16): lane l -> N=l&15, hsel=l>>4; halves 0..7 = K hsel*8+0..7, halves 8..15 = K 16+hsel*8+0..7  (LDS staged as Bt[N][K])

// ---------------- 3x3 conv, implicit GEMM, WMMA ----------------
// loop order: (ky, ic-chunk, kx) so each input row chunk (with +/-1 px halo)
// is staged once in LDS and reused by 3 kx taps.
__global__ __launch_bounds__(256)
void rpn_conv3x3_wmma(const _Float16* __restrict__ X, const _Float16* __restrict__ Wt,
                      const float* __restrict__ bias, _Float16* __restrict__ Y) {
    __shared__ alignas(16) _Float16 As[64][32];   // [oc][k]
    __shared__ alignas(16) _Float16 Bt[66][40];   // [px+1 halo][k] (+pad: 80B row, 16B aligned)

    const int ocb  = blockIdx.x * 64;
    const int y    = blockIdx.y;
    const int b    = blockIdx.z;
    const int tid  = threadIdx.x;
    const int lane = tid & 31;
    const int wave = tid >> 5;
    const int oc_off = (wave >> 1) * 16;
    const int px_off = (wave & 1) * 32;
    const int m    = lane & 15;
    const int hsel = lane >> 4;

    v8f c0 = {}; v8f c1 = {};

    const int a_row = tid >> 2;          // 0..63  (oc within tile)
    const int a_col = (tid & 3) * 8;     // 0,8,16,24
    const int b_k   = tid >> 3;          // 0..31  (ic within chunk)
    const int b_px0 = (tid & 7) * 8;     // 0..56

    for (int ky = 0; ky < 3; ++ky) {
        const int yy = y + ky - 1;
        const bool rowok = (unsigned)yy < (unsigned)Hh;
        for (int icc = 0; icc < 16; ++icc) {
            const int ic0 = icc * 32;

            // ---- stage B row chunk with halo: Bt[px+1][k] = X[b][ic0+k][yy][px]
            if (rowok) {
                const _Float16* srow =
                    X + (((size_t)b * Cin + ic0 + b_k) * Hh + yy) * Ww + b_px0;
                union { uint4 u; _Float16 h[8]; } t;
                t.u = *(const uint4*)srow;
#pragma unroll
                for (int j = 0; j < 8; ++j) Bt[b_px0 + 1 + j][b_k] = t.h[j];
                if (icc + 1 < 16)  // pull next ic chunk's row toward the caches
                    __builtin_prefetch(srow + (size_t)32 * Hh * Ww, 0, 3);
            } else {
#pragma unroll
                for (int j = 0; j < 8; ++j) Bt[b_px0 + 1 + j][b_k] = (_Float16)0.f;
            }
            if (tid < 32) {              // zero halo columns (px = -1 and 64)
                Bt[0][tid]  = (_Float16)0.f;
                Bt[65][tid] = (_Float16)0.f;
            }

            for (int kx = 0; kx < 3; ++kx) {
                // ---- stage A chunk via CDNA5 async global->LDS DMA
                const int pos = ky * 3 + kx;
                async_copy_b128(&As[a_row][a_col],
                                Wt + (size_t)(ocb + a_row) * K9 + pos * Cin + ic0 + a_col);
                async_wait0();
                __syncthreads();   // As ready; Bt ready (dscnt drained at barrier)

                v16h a, b0, b1;
#pragma unroll
                for (int j = 0; j < 8; ++j) {
                    a[j]     = As[oc_off + m][hsel * 8 + j];
                    a[8 + j] = As[oc_off + m][16 + hsel * 8 + j];
                }
#pragma unroll
                for (int j = 0; j < 8; ++j) {
                    b0[j]     = Bt[px_off + kx + m][hsel * 8 + j];
                    b0[8 + j] = Bt[px_off + kx + m][16 + hsel * 8 + j];
                    b1[j]     = Bt[px_off + 16 + kx + m][hsel * 8 + j];
                    b1[8 + j] = Bt[px_off + 16 + kx + m][16 + hsel * 8 + j];
                }
                c0 = __builtin_amdgcn_wmma_f32_16x16x32_f16(false, a, false, b0, (short)0, c0, false, false);
                c1 = __builtin_amdgcn_wmma_f32_16x16x32_f16(false, a, false, b1, (short)0, c1, false, false);
                __syncthreads();   // frag reads done before As/Bt overwrite
            }
        }
    }

    // epilogue: bias + ReLU, store f16
#pragma unroll
    for (int j = 0; j < 8; ++j) {
        int oc = ocb + oc_off + hsel * 8 + j;
        float bse = bias[oc];
        float v0 = c0[j] + bse; v0 = v0 > 0.f ? v0 : 0.f;
        float v1 = c1[j] + bse; v1 = v1 > 0.f ? v1 : 0.f;
        size_t base = (((size_t)b * OC + oc) * Hh + y) * Ww;
        Y[base + px_off + m]      = (_Float16)v0;
        Y[base + px_off + 16 + m] = (_Float16)v1;
    }
}

// ---------------- fused 1x1 convs (cls 18 + bbox 36 padded to 64 rows) ----------------
__global__ __launch_bounds__(256)
void rpn_conv1x1_wmma(const _Float16* __restrict__ C1, const _Float16* __restrict__ W2,
                      const float* __restrict__ bcls, const float* __restrict__ bbbox,
                      float* __restrict__ CLS, float* __restrict__ BBX) {
    __shared__ alignas(16) _Float16 As[64][32];
    __shared__ alignas(16) _Float16 Bt[64][40];

    const int y    = blockIdx.y;
    const int b    = blockIdx.z;
    const int tid  = threadIdx.x;
    const int lane = tid & 31;
    const int wave = tid >> 5;
    const int oc_off = (wave >> 1) * 16;
    const int px_off = (wave & 1) * 32;
    const int m    = lane & 15;
    const int hsel = lane >> 4;

    v8f c0 = {}; v8f c1 = {};

    for (int kc = 0; kc < 16; ++kc) {
        const int ic0 = kc * 32;
        {
            int row = tid >> 2;
            int col = (tid & 3) * 8;
            async_copy_b128(&As[row][col], W2 + (size_t)row * Cin + ic0 + col);
        }
        {
            int k   = tid >> 3;
            int px0 = (tid & 7) * 8;
            union { uint4 u; _Float16 h[8]; } t;
            t.u = *(const uint4*)(C1 + (((size_t)b * Cin + ic0 + k) * Hh + y) * Ww + px0);
#pragma unroll
            for (int j = 0; j < 8; ++j) Bt[px0 + j][k] = t.h[j];
        }
        async_wait0();
        __syncthreads();

        v16h a, b0, b1;
#pragma unroll
        for (int j = 0; j < 8; ++j) {
            a[j]     = As[oc_off + m][hsel * 8 + j];
            a[8 + j] = As[oc_off + m][16 + hsel * 8 + j];
        }
#pragma unroll
        for (int j = 0; j < 8; ++j) {
            b0[j]     = Bt[px_off + m][hsel * 8 + j];
            b0[8 + j] = Bt[px_off + m][16 + hsel * 8 + j];
            b1[j]     = Bt[px_off + 16 + m][hsel * 8 + j];
            b1[8 + j] = Bt[px_off + 16 + m][16 + hsel * 8 + j];
        }
        c0 = __builtin_amdgcn_wmma_f32_16x16x32_f16(false, a, false, b0, (short)0, c0, false, false);
        c1 = __builtin_amdgcn_wmma_f32_16x16x32_f16(false, a, false, b1, (short)0, c1, false, false);
        __syncthreads();
    }

#pragma unroll
    for (int j = 0; j < 8; ++j) {
        int oc = oc_off + hsel * 8 + j;
        if (oc < 18) {
            float bse = bcls[oc];
            size_t base = (((size_t)b * 18 + oc) * Hh + y) * Ww;
            CLS[base + px_off + m]      = c0[j] + bse;
            CLS[base + px_off + 16 + m] = c1[j] + bse;
        } else if (oc < 54) {
            float bse = bbbox[oc - 18];
            size_t base = (((size_t)b * 36 + (oc - 18)) * Hh + y) * Ww;
            BBX[base + px_off + m]      = c0[j] + bse;
            BBX[base + px_off + 16 + m] = c1[j] + bse;
        }
    }
}

// ---------------- scores (softmax fg) + anchor decode + clip ----------------
__global__ void rpn_props(const float* __restrict__ CLS, const float* __restrict__ BBX,
                          const float* __restrict__ im_info,
                          float* __restrict__ SC, float* __restrict__ PR) {
    int gid = blockIdx.x * blockDim.x + threadIdx.x;
    if (gid >= Bn * NPROP) return;
    int a = gid % 9;
    int t = gid / 9;
    int x = t % Ww; t /= Ww;
    int y = t % Hh;
    int b = t / Hh;

    size_t pix = (size_t)y * Ww + x;
    float bg = CLS[(((size_t)b * 18 + a) * Hh * Ww) + pix];
    float fg = CLS[(((size_t)b * 18 + 9 + a) * Hh * Ww) + pix];
    float score = 1.f / (1.f + expf(bg - fg));   // softmax fg prob

    float dx = BBX[(((size_t)b * 36 + a * 4 + 0) * Hh * Ww) + pix];
    float dy = BBX[(((size_t)b * 36 + a * 4 + 1) * Hh * Ww) + pix];
    float dw = BBX[(((size_t)b * 36 + a * 4 + 2) * Hh * Ww) + pix];
    float dh = BBX[(((size_t)b * 36 + a * 4 + 3) * Hh * Ww) + pix];

    float wa = c_AW[a], ha = c_AH[a];
    float cxa = x * 16.f + 8.f, cya = y * 16.f + 8.f;
    float cx = dx * wa + cxa, cy = dy * ha + cya;
    float pw = expf(dw) * wa, ph = expf(dh) * ha;
    float x1 = cx - 0.5f * pw, y1 = cy - 0.5f * ph;
    float x2 = cx + 0.5f * pw, y2 = cy + 0.5f * ph;

    float mx = im_info[b * 3 + 1] - 1.f;
    float my = im_info[b * 3 + 0] - 1.f;
    x1 = fminf(fmaxf(x1, 0.f), mx); x2 = fminf(fmaxf(x2, 0.f), mx);
    y1 = fminf(fmaxf(y1, 0.f), my); y2 = fminf(fmaxf(y2, 0.f), my);

    int idx = ((y * Ww + x) * 9 + a);
    SC[(size_t)b * NPROP + idx] = score;
    float* p = PR + ((size_t)b * NPROP + idx) * 4;
    p[0] = x1; p[1] = y1; p[2] = x2; p[3] = y2;
}

// ---------------- deterministic top-6000 per batch ----------------
__device__ __forceinline__ unsigned rpn_key(float f) {
    unsigned u = __float_as_uint(f);
    return (u & 0x80000000u) ? ~u : (u | 0x80000000u);
}

__global__ __launch_bounds__(1024)
void rpn_topk(const float* __restrict__ SC, const float* __restrict__ PR,
              float* __restrict__ TS, float* __restrict__ TB) {
    const int b = blockIdx.x;
    const int tid = threadIdx.x;
    __shared__ int sh_cnt;
    const float* s = SC + (size_t)b * NPROP;

    // bitwise k-th-largest on sortable keys: largest x with count(key>=x) >= PRE
    unsigned x = 0;
    for (int bit = 31; bit >= 0; --bit) {
        unsigned cand = x | (1u << bit);
        if (tid == 0) sh_cnt = 0;
        __syncthreads();
        int c = 0;
        for (int i = tid; i < NPROP; i += 1024)
            if (rpn_key(s[i]) >= cand) c++;
        atomicAdd(&sh_cnt, c);
        __syncthreads();
        if (sh_cnt >= PRE) x = cand;
        __syncthreads();
    }

    // wave-ordered compaction (deterministic) by first wave only
    if (tid < 32) {
        const int lane = tid;
        const unsigned lt = (1u << lane) - 1u;
        const float* pr = PR + (size_t)b * NPROP * 4;
        float* ts = TS + (size_t)b * PRE;
        float* tb = TB + (size_t)b * PRE * 4;
        int n = 0;
        for (int base = 0; base < NPROP; base += 32) {   // strictly greater first
            int i = base + lane;
            bool p = rpn_key(s[i]) > x;
            unsigned mm = __builtin_amdgcn_ballot_w32(p);
            int pos = n + __popc(mm & lt);
            if (p && pos < PRE) {
                ts[pos] = s[i];
                tb[pos * 4 + 0] = pr[i * 4 + 0]; tb[pos * 4 + 1] = pr[i * 4 + 1];
                tb[pos * 4 + 2] = pr[i * 4 + 2]; tb[pos * 4 + 3] = pr[i * 4 + 3];
            }
            n += __popc(mm);
        }
        for (int base = 0; base < NPROP && n < PRE; base += 32) {  // fill with == threshold
            int i = base + lane;
            bool p = rpn_key(s[i]) == x;
            unsigned mm = __builtin_amdgcn_ballot_w32(p);
            int pos = n + __popc(mm & lt);
            if (p && pos < PRE) {
                ts[pos] = s[i];
                tb[pos * 4 + 0] = pr[i * 4 + 0]; tb[pos * 4 + 1] = pr[i * 4 + 1];
                tb[pos * 4 + 2] = pr[i * 4 + 2]; tb[pos * 4 + 3] = pr[i * 4 + 3];
            }
            n += __popc(mm);
        }
    }
}

// ---------------- greedy NMS, 300 picks, writes (B,300,5) rois ----------------
__global__ __launch_bounds__(256)
void rpn_nms(const float* __restrict__ TS, const float* __restrict__ TB,
             float* __restrict__ out) {
    const int b = blockIdx.x;
    const int tid = threadIdx.x;
    __shared__ float s[PRE];
    __shared__ float area[PRE];
    __shared__ float rs[256];
    __shared__ int   ri[256];
    __shared__ float box_sh[4];
    __shared__ float barea_sh;
    __shared__ int   ok_sh;

    const float* boxes = TB + (size_t)b * PRE * 4;
    const float NEG = -__builtin_inff();

    for (int i = tid; i < PRE; i += 256) {
        s[i] = TS[(size_t)b * PRE + i];
        float x1 = boxes[i * 4], y1 = boxes[i * 4 + 1];
        float x2 = boxes[i * 4 + 2], y2 = boxes[i * 4 + 3];
        area[i] = (x2 - x1 + 1.f) * (y2 - y1 + 1.f);
    }
    __syncthreads();

    for (int it = 0; it < POST; ++it) {
        // argmax with first-index tie-break
        float bs = NEG; int bi = 0x7fffffff;
        for (int i = tid; i < PRE; i += 256) {
            float v = s[i];
            if (v > bs || (v == bs && i < bi)) { bs = v; bi = i; }
        }
        rs[tid] = bs; ri[tid] = bi;
        __syncthreads();
        for (int off = 128; off > 0; off >>= 1) {
            if (tid < off) {
                float v = rs[tid + off]; int j = ri[tid + off];
                if (v > rs[tid] || (v == rs[tid] && j < ri[tid])) { rs[tid] = v; ri[tid] = j; }
            }
            __syncthreads();
        }
        if (tid == 0) {
            int i0 = (ri[0] == 0x7fffffff) ? 0 : ri[0];
            bool ok = rs[0] > NEG;
            ok_sh = ok ? 1 : 0;
            float bx1 = boxes[i0 * 4], by1 = boxes[i0 * 4 + 1];
            float bx2 = boxes[i0 * 4 + 2], by2 = boxes[i0 * 4 + 3];
            box_sh[0] = bx1; box_sh[1] = by1; box_sh[2] = bx2; box_sh[3] = by2;
            barea_sh = area[i0];
            float* o = out + ((size_t)b * POST + it) * 5;
            o[0] = (float)b;
            o[1] = ok ? bx1 : 0.f; o[2] = ok ? by1 : 0.f;
            o[3] = ok ? bx2 : 0.f; o[4] = ok ? by2 : 0.f;
        }
        __syncthreads();
        if (ok_sh) {
            float bx1 = box_sh[0], by1 = box_sh[1], bx2 = box_sh[2], by2 = box_sh[3];
            float ba = barea_sh;
            for (int i = tid; i < PRE; i += 256) {
                if (s[i] == NEG) continue;
                float ix1 = fmaxf(bx1, boxes[i * 4]);
                float iy1 = fmaxf(by1, boxes[i * 4 + 1]);
                float ix2 = fminf(bx2, boxes[i * 4 + 2]);
                float iy2 = fminf(by2, boxes[i * 4 + 3]);
                float iw = fmaxf(ix2 - ix1 + 1.f, 0.f);
                float ih = fmaxf(iy2 - iy1 + 1.f, 0.f);
                float inter = iw * ih;
                float iou = inter / (ba + area[i] - inter);
                if (iou > 0.7f) s[i] = NEG;
            }
        }
        __syncthreads();
    }
}

// ---------------- launch ----------------
extern "C" void kernel_launch(void* const* d_in, const int* in_sizes, int n_in,
                              void* d_out, int out_size, void* d_ws, size_t ws_size,
                              hipStream_t stream) {
    const float* base_feat = (const float*)d_in[0];
    const float* im_info   = (const float*)d_in[1];
    // d_in[2] gt_boxes, d_in[3] num_boxes: unused by the proposal path
    const float* w_conv = (const float*)d_in[4];
    const float* b_conv = (const float*)d_in[5];
    const float* w_cls  = (const float*)d_in[6];
    const float* b_cls  = (const float*)d_in[7];
    const float* w_bbox = (const float*)d_in[8];
    const float* b_bbox = (const float*)d_in[9];

    char* ws = (char*)d_ws;
    size_t off = 0;
    auto arena = [&](size_t bytes) -> char* {
        char* p = ws + off;
        off += (bytes + 255) & ~(size_t)255;
        return p;
    };
    _Float16* X16 = (_Float16*)arena((size_t)Bn * Cin * Hh * Ww * 2);
    _Float16* W16 = (_Float16*)arena((size_t)OC * K9 * 2);
    _Float16* C1  = (_Float16*)arena((size_t)Bn * OC * Hh * Ww * 2);
    _Float16* W2  = (_Float16*)arena((size_t)64 * Cin * 2);
    float* CLS = (float*)arena((size_t)Bn * 18 * Hh * Ww * 4);
    float* BBX = (float*)arena((size_t)Bn * 36 * Hh * Ww * 4);
    float* SC  = (float*)arena((size_t)Bn * NPROP * 4);
    float* PR  = (float*)arena((size_t)Bn * NPROP * 4 * 4);
    float* TS  = (float*)arena((size_t)Bn * PRE * 4);
    float* TB  = (float*)arena((size_t)Bn * PRE * 4 * 4);

    const int nelem = Bn * Cin * Hh * Ww;
    rpn_cvt_input<<<(nelem + 255) / 256, 256, 0, stream>>>(base_feat, X16, nelem);
    rpn_cvt_wconv<<<(OC * K9 + 255) / 256, 256, 0, stream>>>(w_conv, W16);
    rpn_cvt_w2<<<(64 * Cin + 255) / 256, 256, 0, stream>>>(w_cls, w_bbox, W2);

    rpn_conv3x3_wmma<<<dim3(OC / 64, Hh, Bn), 256, 0, stream>>>(X16, W16, b_conv, C1);
    rpn_conv1x1_wmma<<<dim3(1, Hh, Bn), 256, 0, stream>>>(C1, W2, b_cls, b_bbox, CLS, BBX);

    rpn_props<<<(Bn * NPROP + 255) / 256, 256, 0, stream>>>(CLS, BBX, im_info, SC, PR);
    rpn_topk<<<Bn, 1024, 0, stream>>>(SC, PR, TS, TB);
    rpn_nms<<<Bn, 256, 0, stream>>>(TS, TB, (float*)d_out);
}
